// FlashFFN_72138270704295
// MI455X (gfx1250) — compile-verified, matrix operand order checked by reading, not run on hardware
//
#include <hip/hip_runtime.h>

// ---------------------------------------------------------------------------
// FlashFFN for MI455X (gfx1250, wave32, WMMA).
//   h   = gelu(x @ W1 + b1)        [T, D_FF]   bf16 intermediate
//   out = h @ W2 + b2              [T, H]      f32
// bf16 weights (128MB) are L2-resident. GEMM pipeline: ASYNC global->LDS DMA
// (ASYNCcnt, depth-2 double buffer) feeds ds_load fragments + WMMA, so L2
// latency is hidden by a full K-step of compute and costs zero data VGPRs.
// ---------------------------------------------------------------------------

#define T_TOK 4096
#define H_DIM 4096
#define F_DIM 16384

typedef __attribute__((ext_vector_type(16))) __bf16 v16bf;
typedef __attribute__((ext_vector_type(8)))  __bf16 v8bf;
typedef __attribute__((ext_vector_type(8)))  float  v8f;
typedef __attribute__((ext_vector_type(4)))  int    v4i;

typedef __attribute__((address_space(1))) v4i as1_v4i;
typedef __attribute__((address_space(3))) v4i as3_v4i;

static __device__ __forceinline__ unsigned short f32_to_bf16(float f) {
    unsigned int u = __float_as_uint(f);
    unsigned int r = u + 0x7FFFu + ((u >> 16) & 1u);   // round to nearest even
    return (unsigned short)(r >> 16);
}

static __device__ __forceinline__ float gelu_tanh(float v) {
    const float k0 = 0.7978845608028654f;   // sqrt(2/pi)
    const float k1 = 0.044715f;
    return 0.5f * v * (1.0f + tanhf(k0 * (v + k1 * v * v * v)));
}

// Fragment loader (LDS pointers -> ds_load_b128 x2).
// ISA 7.12.2 16-bit layout: chunk0 at half-offset hi*8, chunk1 at hi*8+16.
static __device__ __forceinline__ v16bf load_frag(const unsigned short* p) {
    v8bf lo = *(const v8bf*)(p);
    v8bf hi = *(const v8bf*)(p + 16);
    return __builtin_shufflevector(lo, hi, 0,1,2,3,4,5,6,7,8,9,10,11,12,13,14,15);
}

#define WMMA_BF16(a, b, c) \
    __builtin_amdgcn_wmma_f32_16x16x32_bf16(false, (a), false, (b), (short)0, (c), false, false)

// ---- CDNA5 async global->LDS (16B per lane), tracked with ASYNCcnt --------
static __device__ __forceinline__ void async_cp16(const unsigned short* g, unsigned short* l) {
#if __has_builtin(__builtin_amdgcn_global_load_async_to_lds_b128)
    __builtin_amdgcn_global_load_async_to_lds_b128((as1_v4i*)g, (as3_v4i*)l, 0, 0);
#else
    unsigned loff = (unsigned)(unsigned long long)l;   // LDS addr = low 32 bits
    asm volatile("global_load_async_to_lds_b128 %0, %1, off"
                 :: "v"(loff), "v"(g) : "memory");
#endif
}

#if __has_builtin(__builtin_amdgcn_s_wait_asynccnt)
#define WAIT_ASYNC(n) __builtin_amdgcn_s_wait_asynccnt(n)
#else
#define WAIT_ASYNC(n) asm volatile("s_wait_asynccnt %0" :: "i"(n) : "memory")
#endif

// ---------------------------------------------------------------------------
// Kernel 1: fp32 -> bf16, 4 elements per thread
// ---------------------------------------------------------------------------
__global__ __launch_bounds__(256) void convert_bf16(const float* __restrict__ in,
                                                    unsigned short* __restrict__ out) {
    size_t i = ((size_t)blockIdx.x * 256 + threadIdx.x) * 4;
    float4 v = *(const float4*)(in + i);
    ushort4 o;
    o.x = f32_to_bf16(v.x); o.y = f32_to_bf16(v.y);
    o.z = f32_to_bf16(v.z); o.w = f32_to_bf16(v.w);
    *(ushort4*)(out + i) = o;
}

// ---------------------------------------------------------------------------
// Kernel 2: transpose + convert:  out[c][r] = bf16(in[r][c]),  in is [R][C]
// ---------------------------------------------------------------------------
__global__ __launch_bounds__(256) void transpose_convert(const float* __restrict__ in,
                                                         unsigned short* __restrict__ out,
                                                         int R, int C) {
    __shared__ float tile[32][33];
    const int c0 = blockIdx.x * 32;
    const int r0 = blockIdx.y * 32;
    const int tx = threadIdx.x, ty = threadIdx.y;
#pragma unroll
    for (int i = 0; i < 32; i += 8)
        tile[ty + i][tx] = in[(size_t)(r0 + ty + i) * C + c0 + tx];
    __syncthreads();
#pragma unroll
    for (int i = 0; i < 32; i += 8)
        out[(size_t)(c0 + ty + i) * R + r0 + tx] = f32_to_bf16(tile[tx][ty + i]);
}

// ---------------------------------------------------------------------------
// Kernel 3: bf16 WMMA GEMM, async double-buffered LDS pipeline.
//   Block: 256 thr (8 waves 2x4), block tile 128x256, wave tile 64x64.
//   K step 32. LDS rows padded to 40 halves (80B) -> conflict-free b128.
// ---------------------------------------------------------------------------
#define LDSPITCH 40

template <int GELU_OUT>
__global__ __launch_bounds__(256) void ffn_gemm(const unsigned short* __restrict__ A,
                                                const unsigned short* __restrict__ Bt,
                                                const float* __restrict__ bias,
                                                void* __restrict__ outp,
                                                int M, int N, int K) {
    __shared__ unsigned short sA[2][128][LDSPITCH];   // 20 KB
    __shared__ unsigned short sB[2][256][LDSPITCH];   // 40 KB

    const int t     = threadIdx.x;
    const int lane  = t & 31;
    const int wave  = t >> 5;
    const int waveM = wave >> 2;          // 0..1
    const int waveN = wave & 3;           // 0..3
    const int m_blk = blockIdx.x * 128;
    const int n_blk = blockIdx.y * 256;

    const int row = lane & 15;
    const int hi  = lane >> 4;

    // ---- cooperative staging addresses: thread t moves 6 x 16B per tile ----
    const int sr = t >> 2;                // 0..63  (tile row)
    const int sc = (t & 3) * 8;           // half offset 0/8/16/24
    const unsigned short* gA = A  + (size_t)(m_blk + sr) * K + sc;
    const unsigned short* gB = Bt + (size_t)(n_blk + sr) * K + sc;

    // ---- fragment base rows in LDS ----
    const int afr = waveM * 64 + row;     // + i*16
    const int bfr = waveN * 64 + row;     // + j*16

    v8f acc[4][4];
#pragma unroll
    for (int i = 0; i < 4; ++i)
#pragma unroll
        for (int j = 0; j < 4; ++j) acc[i][j] = (v8f)(0.0f);

    const int KT = K >> 5;

    // ---- prologue: async-stage tiles 0 and 1 ----
    {
        async_cp16(gA,                   &sA[0][sr      ][sc]);
        async_cp16(gA + (size_t)64 * K,  &sA[0][sr +  64][sc]);
        async_cp16(gB,                   &sB[0][sr      ][sc]);
        async_cp16(gB + (size_t)64 * K,  &sB[0][sr +  64][sc]);
        async_cp16(gB + (size_t)128 * K, &sB[0][sr + 128][sc]);
        async_cp16(gB + (size_t)192 * K, &sB[0][sr + 192][sc]);
        const size_t k1 = (KT > 1) ? 32 : 0;
        async_cp16(gA + k1,                   &sA[1][sr      ][sc]);
        async_cp16(gA + k1 + (size_t)64 * K,  &sA[1][sr +  64][sc]);
        async_cp16(gB + k1,                   &sB[1][sr      ][sc]);
        async_cp16(gB + k1 + (size_t)64 * K,  &sB[1][sr +  64][sc]);
        async_cp16(gB + k1 + (size_t)128 * K, &sB[1][sr + 128][sc]);
        async_cp16(gB + k1 + (size_t)192 * K, &sB[1][sr + 192][sc]);
    }

    for (int kt = 0; kt < KT; ++kt) {
        const int buf = kt & 1;
        // Oldest 6 asyncs (tile kt) done when <=6 remain (per-wave, in-order).
        WAIT_ASYNC(6);
        __syncthreads();                  // tile kt visible from all waves

        v16bf a[4];
#pragma unroll
        for (int i = 0; i < 4; ++i)
            a[i] = load_frag(&sA[buf][afr + i * 16][hi * 8]);
#pragma unroll
        for (int j = 0; j < 4; ++j) {
            v16bf b = load_frag(&sB[buf][bfr + j * 16][hi * 8]);
#pragma unroll
            for (int i = 0; i < 4; ++i)
                acc[i][j] = WMMA_BF16(a[i], b, acc[i][j]);
        }

        __syncthreads();                  // all waves done reading buf

        // refill the just-freed buffer with tile kt+2 (clamped at the tail)
        const size_t koff = (size_t)((kt + 2 < KT) ? (kt + 2) : 0) << 5;
        async_cp16(gA + koff,                   &sA[buf][sr      ][sc]);
        async_cp16(gA + koff + (size_t)64 * K,  &sA[buf][sr +  64][sc]);
        async_cp16(gB + koff,                   &sB[buf][sr      ][sc]);
        async_cp16(gB + koff + (size_t)64 * K,  &sB[buf][sr +  64][sc]);
        async_cp16(gB + koff + (size_t)128 * K, &sB[buf][sr + 128][sc]);
        async_cp16(gB + koff + (size_t)192 * K, &sB[buf][sr + 192][sc]);
    }

    // Epilogue. C/D layout: col = base + (lane&15); row = base + (lane>>4)*8 + i
    const int m_base = m_blk + waveM * 64;
    const int n_base = n_blk + waveN * 64;
    const int col_off = lane & 15;
    const int row_off = hi * 8;
#pragma unroll
    for (int mt = 0; mt < 4; ++mt) {
#pragma unroll
        for (int nt = 0; nt < 4; ++nt) {
            const int col = n_base + nt * 16 + col_off;
            const float bv = bias[col];
#pragma unroll
            for (int i = 0; i < 8; ++i) {
                const int r = m_base + mt * 16 + row_off + i;
                float v = acc[mt][nt][i] + bv;
                if (GELU_OUT) {
                    ((unsigned short*)outp)[(size_t)r * N + col] = f32_to_bf16(gelu_tanh(v));
                } else {
                    ((float*)outp)[(size_t)r * N + col] = v;
                }
            }
        }
    }
}

// ---------------------------------------------------------------------------
extern "C" void kernel_launch(void* const* d_in, const int* in_sizes, int n_in,
                              void* d_out, int out_size, void* d_ws, size_t ws_size,
                              hipStream_t stream) {
    const float* x  = (const float*)d_in[0];   // [T, H]
    const float* W1 = (const float*)d_in[1];   // [H, F]
    const float* b1 = (const float*)d_in[2];   // [F]
    const float* W2 = (const float*)d_in[3];   // [F, H]
    const float* b2 = (const float*)d_in[4];   // [H]
    float* out = (float*)d_out;                // [T, H]

    const size_t XB_BYTES  = (size_t)T_TOK * H_DIM * 2;   //  32 MB
    const size_t W1T_BYTES = (size_t)F_DIM * H_DIM * 2;   // 128 MB
    const size_t W2T_BYTES = (size_t)H_DIM * F_DIM * 2;   // 128 MB

    char* ws = (char*)d_ws;
    unsigned short* Xb  = (unsigned short*)(ws);                                   // [T][H]
    unsigned short* W1t = (unsigned short*)(ws + XB_BYTES);                        // [F][H]
    unsigned short* W2t = (unsigned short*)(ws + XB_BYTES + W1T_BYTES);            // [H][F]
    unsigned short* Hb  = (unsigned short*)(ws + XB_BYTES + W1T_BYTES + W2T_BYTES);// [T][F]

    // 1) activations -> bf16
    {
        size_t n = (size_t)T_TOK * H_DIM;                 // 16M
        convert_bf16<<<dim3((unsigned)(n / (256 * 4))), dim3(256), 0, stream>>>(x, Xb);
    }
    // 2) weights -> bf16, transposed (K-major)
    transpose_convert<<<dim3(F_DIM / 32, H_DIM / 32), dim3(32, 8), 0, stream>>>(W1, W1t, H_DIM, F_DIM);
    transpose_convert<<<dim3(H_DIM / 32, F_DIM / 32), dim3(32, 8), 0, stream>>>(W2, W2t, F_DIM, H_DIM);
    // 3) h = gelu(x @ W1 + b1)  -> bf16 [T, F]
    ffn_gemm<1><<<dim3(T_TOK / 128, F_DIM / 256), dim3(256), 0, stream>>>(
        Xb, W1t, b1, (void*)Hb, T_TOK, F_DIM, H_DIM);
    // 4) out = h @ W2 + b2  -> f32 [T, H]
    ffn_gemm<0><<<dim3(T_TOK / 128, H_DIM / 256), dim3(256), 0, stream>>>(
        Hb, W2t, b2, (void*)out, T_TOK, H_DIM, F_DIM);
}
